// MultiScaleRetention_9698036154630
// MI455X (gfx1250) — compile-verified, matrix operand order checked by reading, not run on hardware
//
#include <hip/hip_runtime.h>
#include <hip/hip_bf16.h>

#define B_  2
#define T_  2048
#define E_  1024
#define V_  2048
#define NH_ 8
#define KD_ 128
#define HD_ 256
#define EPS_ 1e-6f

typedef __attribute__((ext_vector_type(16))) __bf16 v16bf;
typedef __attribute__((ext_vector_type(8)))  __bf16 v8bf;
typedef __attribute__((ext_vector_type(8)))  float  v8f;
typedef __attribute__((ext_vector_type(4)))  unsigned u32x4;
typedef __attribute__((ext_vector_type(4)))  int      i32x4;
typedef __attribute__((ext_vector_type(8)))  int      i32x8;

#if __has_builtin(__builtin_amdgcn_tensor_load_to_lds)
#define USE_TDM 1
#else
#define USE_TDM 0
#endif

// GEMM epilogue modes
#define OUT_F32   0
#define OUT_BF16  1
#define OUT_THETA 2   // rotary theta shift + scale, bf16 out

__device__ __forceinline__ __bf16 f2bf(float f) {
  unsigned u = __builtin_bit_cast(unsigned, f);
  u += 0x7FFFu + ((u >> 16) & 1u);          // round-to-nearest-even
  unsigned short h = (unsigned short)(u >> 16);
  return __builtin_bit_cast(__bf16, h);
}
__device__ __forceinline__ float bf2f(__bf16 b) {
  unsigned short h = __builtin_bit_cast(unsigned short, b);
  unsigned u = ((unsigned)h) << 16;
  return __builtin_bit_cast(float, u);
}
__device__ __forceinline__ v8f v8f_zero() {
  v8f z;
  #pragma unroll
  for (int i = 0; i < 8; ++i) z[i] = 0.f;
  return z;
}
// Build a v16bf A/B operand from two 16-byte halves per ISA 16-bit layout:
// lanes 0-15: K[0..7] + K[16..23]; lanes 16-31: K[8..15] + K[24..31]
__device__ __forceinline__ v16bf pack_op(const __bf16* row, int koff) {
  v8bf lo = *(const v8bf*)(row + koff);
  v8bf hi = *(const v8bf*)(row + 16 + koff);
  v16bf r;
  #pragma unroll
  for (int j = 0; j < 8; ++j) { r[j] = lo[j]; r[j + 8] = hi[j]; }
  return r;
}

#if USE_TDM
// Issue one TDM 2D tile load (global -> LDS, row-major, bf16 elements).
// tile: rows x rowElems, global row stride = rowStrideElems. Tracked by TENSORcnt.
__device__ __forceinline__ void tdm_load_2d(unsigned ldsAddr, const void* gptr,
                                            unsigned rows, unsigned rowElems,
                                            unsigned long long rowStrideElems) {
  unsigned long long ga = (unsigned long long)gptr;
  u32x4 g0;
  g0[0] = 1u;                                               // count=1 (valid user D#)
  g0[1] = ldsAddr;                                          // lds_addr (bytes)
  g0[2] = (unsigned)ga;                                     // global_addr[31:0]
  g0[3] = ((unsigned)(ga >> 32) & 0x01FFFFFFu) | (2u << 30); // global_addr[56:32] + type=2
  i32x8 g1;
  g1[0] = (int)(1u << 16);                                  // data_size=1 (2 bytes)
  g1[1] = 0;                                                // tensor_dim0=1<<30 -> lo16=0
  g1[2] = (int)0x4000u;                                     // tensor_dim0 hi16; tensor_dim1 lo16=0
  g1[3] = (int)(0x4000u | (rowElems << 16));                // tensor_dim1 hi16; tile_dim0
  g1[4] = (int)(rows & 0xFFFFu);                            // tile_dim1; tile_dim2=0
  g1[5] = (int)(unsigned)rowStrideElems;                    // tensor_dim0_stride[31:0]
  g1[6] = (int)((unsigned)(rowStrideElems >> 32) & 0xFFFFu);// stride hi16; dim1_stride lo16=0
  g1[7] = 0;
  i32x4 z4 = {0, 0, 0, 0};
#if defined(__clang_major__) && __clang_major__ >= 23
  i32x8 z8 = {0, 0, 0, 0, 0, 0, 0, 0};
  __builtin_amdgcn_tensor_load_to_lds(g0, g1, z4, z4, z8, 0);
#else
  __builtin_amdgcn_tensor_load_to_lds(g0, g1, z4, z4, 0);
#endif
}
#else
// Fallback: cooperative 128x32 bf16 tile copy by all 256 threads.
__device__ __forceinline__ void coop_copy_tile(__bf16* lds, const __bf16* g,
                                               long long strideElems, int tid) {
  #pragma unroll
  for (int it = 0; it < 2; ++it) {
    int chunk = it * 256 + tid;          // 0..511
    int row = chunk >> 2;                // 0..127
    int c8  = (chunk & 3) * 8;           // 0,8,16,24
    *(v8bf*)(lds + row * 32 + c8) = *(const v8bf*)(g + (long long)row * strideElems + c8);
  }
}
#endif

// -------------------------------------------- vectorized f32 -> bf16 convert
__global__ void k_f32_to_bf16(const float* __restrict__ in, __bf16* __restrict__ out, int n8) {
  int i = blockIdx.x * blockDim.x + threadIdx.x;   // one thread = 8 elements
  if (i < n8) {
    const float4* p = (const float4*)in + (size_t)i * 2;
    float4 a = p[0], bq = p[1];
    v8bf o;
    o[0] = f2bf(a.x);  o[1] = f2bf(a.y);  o[2] = f2bf(a.z);  o[3] = f2bf(a.w);
    o[4] = f2bf(bq.x); o[5] = f2bf(bq.y); o[6] = f2bf(bq.z); o[7] = f2bf(bq.w);
    ((v8bf*)out)[i] = o;
  }
}

// ---------------- GEMM: C[M,N] = A[M,K] @ W[N,K]^T (bf16 WMMA, TDM-staged LDS)
// block 256 (8 waves) -> 128x128 tile; waves 4x2, each 32x64 (2x4 wmma tiles).
// Double-buffered 128x32 A/W tiles staged by the Tensor Data Mover.
// MODE=OUT_THETA fuses the rotary theta shift: the rotate partner x[d^1] for
// output (m,n) lives in lane^1 of the same accumulator VGPR (d&1 == lane&1).
template <int MODE>
__global__ void __launch_bounds__(256)
k_gemm(const __bf16* __restrict__ A, const __bf16* __restrict__ W,
       void* __restrict__ Cout, const float* __restrict__ sin_,
       const float* __restrict__ cos_, float scale, int M, int N, int K) {
  (void)M;
  __shared__ __bf16 AS[2][128 * 32];   // 8 KB each
  __shared__ __bf16 WS[2][128 * 32];

  const int tid  = threadIdx.x;
  const int wave = tid >> 5;
  const int lane = tid & 31;
  const int half = lane >> 4;
  const int lm   = lane & 15;
  const int koff = half ? 8 : 0;
  const int wm   = wave & 3;           // row sub-tile (32 rows)
  const int wn   = wave >> 2;          // col sub-tile (64 cols)
  const int m0 = blockIdx.y * 128;
  const int n0 = blockIdx.x * 128;

#if USE_TDM
  const bool w0 = (__builtin_amdgcn_readfirstlane((int)threadIdx.x) >> 5) == 0;
#endif

  v8f acc[2][4];
  #pragma unroll
  for (int i = 0; i < 2; ++i)
    #pragma unroll
    for (int j = 0; j < 4; ++j) acc[i][j] = v8f_zero();

  // prologue: stage buffer 0
#if USE_TDM
  if (w0) {
    tdm_load_2d((unsigned)(unsigned long long)&AS[0][0], A + (size_t)m0 * K, 128, 32, (unsigned long long)K);
    tdm_load_2d((unsigned)(unsigned long long)&WS[0][0], W + (size_t)n0 * K, 128, 32, (unsigned long long)K);
  }
  __builtin_amdgcn_s_wait_tensorcnt(0);
#else
  coop_copy_tile(&AS[0][0], A + (size_t)m0 * K, K, tid);
  coop_copy_tile(&WS[0][0], W + (size_t)n0 * K, K, tid);
#endif
  __syncthreads();

  int p = 0;
  for (int kc = 0; kc < K; kc += 32) {
    // stage next buffer while computing current
    if (kc + 32 < K) {
#if USE_TDM
      if (w0) {
        tdm_load_2d((unsigned)(unsigned long long)&AS[p ^ 1][0], A + (size_t)m0 * K + kc + 32, 128, 32, (unsigned long long)K);
        tdm_load_2d((unsigned)(unsigned long long)&WS[p ^ 1][0], W + (size_t)n0 * K + kc + 32, 128, 32, (unsigned long long)K);
      }
#else
      coop_copy_tile(&AS[p ^ 1][0], A + (size_t)m0 * K + kc + 32, K, tid);
      coop_copy_tile(&WS[p ^ 1][0], W + (size_t)n0 * K + kc + 32, K, tid);
#endif
    }

    const __bf16* as = &AS[p][0];
    const __bf16* ws = &WS[p][0];
    v16bf a[2], b[4];
    #pragma unroll
    for (int r = 0; r < 2; ++r) a[r] = pack_op(as + (wm * 32 + r * 16 + lm) * 32, koff);
    #pragma unroll
    for (int c = 0; c < 4; ++c) b[c] = pack_op(ws + (wn * 64 + c * 16 + lm) * 32, koff);
    #pragma unroll
    for (int ti = 0; ti < 2; ++ti)
      #pragma unroll
      for (int tj = 0; tj < 4; ++tj)
        acc[ti][tj] = __builtin_amdgcn_wmma_f32_16x16x32_bf16(
            false, a[ti], false, b[tj], (short)0, acc[ti][tj], false, false);

    __syncthreads();                        // all waves done reading buffer p
#if USE_TDM
    __builtin_amdgcn_s_wait_tensorcnt(0);   // next buffer landed (no-op for non-issuers)
#endif
    __syncthreads();                        // release readers into next buffer
    p ^= 1;
  }

  #pragma unroll
  for (int ti = 0; ti < 2; ++ti)
    #pragma unroll
    for (int tj = 0; tj < 4; ++tj)
      #pragma unroll
      for (int j = 0; j < 8; ++j) {
        size_t m = m0 + wm * 32 + ti * 16 + j + 8 * half;
        size_t n = n0 + wn * 64 + tj * 16 + lm;
        float v = acc[ti][tj][j];
        if (MODE == OUT_F32) {
          ((float*)Cout)[m * N + n] = v;
        } else if (MODE == OUT_BF16) {
          ((__bf16*)Cout)[m * N + n] = f2bf(v);
        } else {  // OUT_THETA
          float partner = __shfl_xor(v, 1, 32);         // x[d^1]
          int d = (int)(n & (KD_ - 1));
          float rot = (d & 1) ? partner : -partner;     // rotate_every_two
          int t = (int)(m & (T_ - 1));
          float s = sin_[t * KD_ + d], c = cos_[t * KD_ + d];
          ((__bf16*)Cout)[m * N + n] = f2bf((v * c + rot * s) * scale);
        }
      }
}

// ---------------------------------------------------- fused retention kernel
// grid (T/64, NH, B), block 128 (4 waves); each wave: 16 query rows x HD cols.
// Rowsum partials stay per-lane; reduced once in the epilogue. Epilogue also
// fuses denom clip + RMS norm + SiLU gate, writing x = silu(g)*ro directly.
__global__ void __launch_bounds__(128)
k_attn(const __bf16* __restrict__ qr, const __bf16* __restrict__ kr,
       const __bf16* __restrict__ vb, const float* __restrict__ mask,
       const float* __restrict__ gf, __bf16* __restrict__ xb) {
  __shared__ __bf16 vT[HD_ * 32];        // [d][s] transposed v tile, 16 KB
  __shared__ __bf16 qkS[4 * 16 * 32];    // per-wave 16x32 qk staging, 4 KB

  const int tid  = threadIdx.x;
  const int wave = tid >> 5;
  const int lane = tid & 31;
  const int half = lane >> 4;
  const int lm   = lane & 15;
  const int koff = half ? 8 : 0;

  const int t0 = blockIdx.x * 64 + wave * 16;
  const int h  = blockIdx.y;
  const int b  = blockIdx.z;

  // preload qr A-operand chunks (16 rows x 128 head dims = 4 K-chunks)
  const __bf16* qbase = qr + ((size_t)(b * T_ + t0 + lm)) * E_ + h * KD_;
  v16bf aq[4];
  #pragma unroll
  for (int kc = 0; kc < 4; ++kc) aq[kc] = pack_op(qbase + kc * 32, koff);

  v8f acc[16];
  #pragma unroll
  for (int i = 0; i < 16; ++i) acc[i] = v8f_zero();
  float rs[8] = {0.f, 0.f, 0.f, 0.f, 0.f, 0.f, 0.f, 0.f};  // per-lane column partials
  __bf16* qkW = qkS + wave * 512;

  for (int s0 = 0; s0 < T_; s0 += 32) {
    __syncthreads();
    // cooperative transposed v-tile staging (32 keys x 256 head dims)
    #pragma unroll
    for (int it = 0; it < 8; ++it) {
      int chunk = it * 128 + tid;              // 0..1023
      int s  = chunk >> 5;
      int d8 = (chunk & 31) * 8;
      v8bf vv = *(const v8bf*)(vb + ((size_t)(b * T_ + s0 + s)) * V_ + h * HD_ + d8);
      #pragma unroll
      for (int j = 0; j < 8; ++j) vT[(d8 + j) * 32 + s] = vv[j];
    }
    __syncthreads();

    // stage 1: qk(16x32) = qr @ kr^T, masked; per-lane rowsum partials; stage to LDS
    #pragma unroll
    for (int nt = 0; nt < 2; ++nt) {
      v8f qk = v8f_zero();
      const __bf16* kbase = kr + ((size_t)(b * T_ + s0 + nt * 16 + lm)) * E_ + h * KD_;
      #pragma unroll
      for (int kc = 0; kc < 4; ++kc) {
        v16bf bk = pack_op(kbase + kc * 32, koff);
        qk = __builtin_amdgcn_wmma_f32_16x16x32_bf16(
            false, aq[kc], false, bk, (short)0, qk, false, false);
      }
      #pragma unroll
      for (int j = 0; j < 8; ++j) {
        int m = j + 8 * half;
        float mv = mask[((size_t)h * T_ + (t0 + m)) * T_ + s0 + nt * 16 + lm];
        float qv = qk[j] * mv;
        rs[j] += qv;                           // deferred reduction
        qkW[m * 32 + nt * 16 + lm] = f2bf(qv);
      }
    }

    // stage 2: acc(16x256) += qk(16x32) @ v(32x256)
    v16bf aqk = pack_op(qkW + lm * 32, koff);
    #pragma unroll
    for (int nt2 = 0; nt2 < 16; ++nt2) {
      v16bf bv = pack_op(vT + (nt2 * 16 + lm) * 32, koff);
      acc[nt2] = __builtin_amdgcn_wmma_f32_16x16x32_bf16(
          false, aqk, false, bv, (short)0, acc[nt2], false, false);
    }
  }

  // epilogue: one-shot rowsum reduction, denom clip + RMS norm, fused SiLU gate
  float scl[8];
  #pragma unroll
  for (int j = 0; j < 8; ++j) {
    float rsum = rs[j];
    #pragma unroll
    for (int off = 1; off < 16; off <<= 1) rsum += __shfl_xor(rsum, off, 32);
    float denom = fabsf(rsum);
    denom = denom < 1.f ? 1.f : denom;
    float inv = 1.f / denom;
    float ss = 0.f;
    #pragma unroll
    for (int nt2 = 0; nt2 < 16; ++nt2) { float v = acc[nt2][j] * inv; ss += v * v; }
    #pragma unroll
    for (int off = 1; off < 16; off <<= 1) ss += __shfl_xor(ss, off, 32);
    scl[j] = inv * rsqrtf(ss * (1.0f / HD_) + EPS_);
  }
  #pragma unroll
  for (int nt2 = 0; nt2 < 16; ++nt2)
    #pragma unroll
    for (int j = 0; j < 8; ++j) {
      int m = j + 8 * half;
      size_t idx = ((size_t)(b * T_ + t0 + m)) * V_ + h * HD_ + nt2 * 16 + lm;
      float gv = gf[idx];
      float sg = gv / (1.f + __expf(-gv));    // silu
      xb[idx] = f2bf(sg * acc[nt2][j] * scl[j]);
    }
}

// ---------------------------------------------------------------- launcher
extern "C" void kernel_launch(void* const* d_in, const int* in_sizes, int n_in,
                              void* d_out, int out_size, void* d_ws, size_t ws_size,
                              hipStream_t stream) {
  (void)in_sizes; (void)n_in; (void)out_size; (void)ws_size;
  const float* hs   = (const float*)d_in[0];
  const float* sinp = (const float*)d_in[1];
  const float* cosp = (const float*)d_in[2];
  const float* mask = (const float*)d_in[3];
  const float* Wq   = (const float*)d_in[4];
  const float* Wk   = (const float*)d_in[5];
  const float* Wv   = (const float*)d_in[6];
  const float* Wg   = (const float*)d_in[7];
  const float* Wo   = (const float*)d_in[8];
  float* out = (float*)d_out;

  char* ws = (char*)d_ws;
  size_t off = 0;
  auto alloc = [&](size_t bytes) -> void* {
    void* p = ws + off;
    off += (bytes + 255) & ~(size_t)255;
    return p;
  };
  const size_t BT = (size_t)B_ * T_;

  __bf16* hsb = (__bf16*)alloc(BT * E_ * 2);
  __bf16* Wqb = (__bf16*)alloc((size_t)E_ * E_ * 2);
  __bf16* Wkb = (__bf16*)alloc((size_t)E_ * E_ * 2);
  __bf16* Wvb = (__bf16*)alloc((size_t)V_ * E_ * 2);
  __bf16* Wgb = (__bf16*)alloc((size_t)V_ * E_ * 2);
  __bf16* Wob = (__bf16*)alloc((size_t)E_ * V_ * 2);
  float*  gf  = (float*)alloc(BT * V_ * 4);
  __bf16* qrb = (__bf16*)alloc(BT * E_ * 2);
  __bf16* krb = (__bf16*)alloc(BT * E_ * 2);
  __bf16* vbb = (__bf16*)alloc(BT * V_ * 2);
  __bf16* xb  = (__bf16*)alloc(BT * V_ * 2);

  auto cvt = [&](const float* src, __bf16* dst, size_t n) {
    k_f32_to_bf16<<<dim3((unsigned)(n / 2048)), dim3(256), 0, stream>>>(src, dst, (int)(n / 8));
  };
  cvt(hs, hsb, BT * E_);
  cvt(Wq, Wqb, (size_t)E_ * E_);
  cvt(Wk, Wkb, (size_t)E_ * E_);
  cvt(Wv, Wvb, (size_t)V_ * E_);
  cvt(Wg, Wgb, (size_t)V_ * E_);
  cvt(Wo, Wob, (size_t)E_ * V_);

  dim3 gblk(256);
  const float kscale = 0.08838834764831845f;   // (E/NH)^-0.5
  // q/k projections with fused rotary shift -> bf16
  k_gemm<OUT_THETA><<<dim3(E_ / 128, (unsigned)(BT / 128)), gblk, 0, stream>>>(
      hsb, Wqb, qrb, sinp, cosp, 1.0f,   (int)BT, E_, E_);
  k_gemm<OUT_THETA><<<dim3(E_ / 128, (unsigned)(BT / 128)), gblk, 0, stream>>>(
      hsb, Wkb, krb, sinp, cosp, kscale, (int)BT, E_, E_);
  // v projection -> bf16 ; g projection -> f32
  k_gemm<OUT_BF16><<<dim3(V_ / 128, (unsigned)(BT / 128)), gblk, 0, stream>>>(
      hsb, Wvb, vbb, nullptr, nullptr, 1.0f, (int)BT, V_, E_);
  k_gemm<OUT_F32><<<dim3(V_ / 128, (unsigned)(BT / 128)), gblk, 0, stream>>>(
      hsb, Wgb, gf, nullptr, nullptr, 1.0f, (int)BT, V_, E_);

  // fused retention (+ denom clip + RMS norm + SiLU gate)
  k_attn<<<dim3(T_ / 64, NH_, B_), dim3(128), 0, stream>>>(qrb, krb, vbb, mask, gf, xb);

  // output projection -> fp32 out
  k_gemm<OUT_F32><<<dim3(E_ / 128, (unsigned)(BT / 128)), gblk, 0, stream>>>(
      xb, Wob, out, nullptr, nullptr, 1.0f, (int)BT, E_, V_);
}
